// WinEncoderTransformer_29377576305191
// MI455X (gfx1250) — matrix operand order repaired and checked by reading, not compiled, not run on hardware
//
#include <hip/hip_runtime.h>
#include <hip/hip_bf16.h>
#include <math.h>
#include <stdint.h>

#define B_    2
#define C_    256
#define H_    128
#define W_    128
#define TOK_  (B_*H_*W_)
#define NHEAD_ 8
#define HDIM_ 32
#define DFF_  512
#define L_    4

// gfx1250 async global->LDS staging via inline asm (flip to 0 if assembler rejects)
#define USE_ASYNC_LDS 1

typedef __attribute__((ext_vector_type(16))) _Float16 v16h;
typedef __attribute__((ext_vector_type(8)))  float    v8f;

#define EPI_BIAS    0
#define EPI_GELU    1
#define EPI_RES_WIN 2
#define EPI_RES_ADD 3

__device__ __forceinline__ v8f wmma_f16(v16h a, v16h b, v8f c) {
  // (neg_a, A, neg_b, B, c_mod, C, reuse_a, reuse_b)
  return __builtin_amdgcn_wmma_f32_16x16x32_f16(false, a, false, b, (short)0, c, false, false);
}

__device__ __forceinline__ v16h cvt16(float4 a, float4 b, float4 c, float4 d) {
  v16h v;
  v[0]=(_Float16)a.x;  v[1]=(_Float16)a.y;  v[2]=(_Float16)a.z;  v[3]=(_Float16)a.w;
  v[4]=(_Float16)b.x;  v[5]=(_Float16)b.y;  v[6]=(_Float16)b.z;  v[7]=(_Float16)b.w;
  v[8]=(_Float16)c.x;  v[9]=(_Float16)c.y;  v[10]=(_Float16)c.z; v[11]=(_Float16)c.w;
  v[12]=(_Float16)d.x; v[13]=(_Float16)d.y; v[14]=(_Float16)d.z; v[15]=(_Float16)d.w;
  return v;
}

// ---------------------------------------------------------------------------
// WMMA GEMM, 64x64 tile per 128-thread block (4 waves):
//   Out[M,Ncols] = A[M,K] * W[Ncols,K]^T (+epi), f32 in, f16 WMMA, f32 accum.
// Weight panel (64 cols x 32 k) is converted to f16 and staged in LDS each
// k-step, shared by all 4 waves; each wave reuses its A operand across the
// 4 column tiles (4 v_wmma per step).
// Operand layouts used by this code (contiguous k, same for A and B):
//   A 16x32 f16 : lane m=lane&15, hx=lane>>4; slot s -> k = hx*16 + s
//   B 32x16 f16 : lane n=lane&15, hx=lane>>4; slot s -> k = hx*16 + s
//   C/D 16x16 f32: vgpr r, lane l -> (M = r + 8*(l>>4), N = l&15)
// Requires M % 64 == 0, Ncols % 64 == 0, K % 32 == 0 (true for all calls).
// ---------------------------------------------------------------------------
__global__ __launch_bounds__(128) void gemm64_kernel(
    const float* __restrict__ A, int lda,
    const float* __restrict__ Wt, int ldw,
    const float* __restrict__ bias,
    float* __restrict__ Out, int ldo,
    int M, int Ncols, int K, int epi,
    int wh, int ww, int nWh, int nWw)
{
  __shared__ _Float16 Bl[64 * 32];           // [col][k] panel, 4 KB

  const int lane = threadIdx.x & 31;
  const int warp = threadIdx.x >> 5;
  const int nbx  = Ncols >> 6;
  const int bm   = blockIdx.x / nbx;
  const int bn   = blockIdx.x % nbx;
  const int row0 = (bm << 6) + (warp << 4);  // 16-row slice per wave
  const int col0 = bn << 6;
  const int m  = lane & 15;
  const int hx = lane >> 4;
  const int n  = m;

  // staging role: thread -> (col, k-half)
  const int scol = threadIdx.x >> 1;
  const int skh  = threadIdx.x & 1;
  const float* wsrc0 = Wt + (long)(col0 + scol) * ldw + skh * 16;
  const float* arow  = A  + (long)(row0 + m) * lda + hx * 16;

  v8f acc0 = {}, acc1 = {}, acc2 = {}, acc3 = {};

  for (int k0 = 0; k0 < K; k0 += 32) {
    // ---- cooperative weight-panel stage (global f32 -> LDS f16) ----
    {
      const float* ws = wsrc0 + k0;
      float4 w0 = *(const float4*)(ws);
      float4 w1 = *(const float4*)(ws + 4);
      float4 w2 = *(const float4*)(ws + 8);
      float4 w3 = *(const float4*)(ws + 12);
      *(v16h*)&Bl[scol * 32 + skh * 16] = cvt16(w0, w1, w2, w3);
    }
    __syncthreads();

    // ---- A operand (per-wave, reused across 4 column tiles) ----
    const float* pa = arow + k0;
    float4 a0 = *(const float4*)(pa);
    float4 a1 = *(const float4*)(pa + 4);
    float4 a2 = *(const float4*)(pa + 8);
    float4 a3 = *(const float4*)(pa + 12);
    v16h av = cvt16(a0, a1, a2, a3);

    v16h b0 = *(const v16h*)&Bl[(0 * 16 + n) * 32 + hx * 16];
    v16h b1 = *(const v16h*)&Bl[(1 * 16 + n) * 32 + hx * 16];
    v16h b2 = *(const v16h*)&Bl[(2 * 16 + n) * 32 + hx * 16];
    v16h b3 = *(const v16h*)&Bl[(3 * 16 + n) * 32 + hx * 16];
    acc0 = wmma_f16(av, b0, acc0);
    acc1 = wmma_f16(av, b1, acc1);
    acc2 = wmma_f16(av, b2, acc2);
    acc3 = wmma_f16(av, b3, acc3);
    __syncthreads();
  }

  // ---- epilogue over the 4 column tiles ----
  v8f accs[4] = {acc0, acc1, acc2, acc3};
  #pragma unroll
  for (int tn = 0; tn < 4; ++tn) {
    const int cn = col0 + tn * 16 + n;
    const float bval = bias ? bias[cn] : 0.0f;
    #pragma unroll
    for (int r = 0; r < 8; ++r) {
      const int mm = row0 + r + 8 * hx;
      float v = accs[tn][r] + bval;
      if (epi == EPI_BIAS) {
        Out[(long)mm * ldo + cn] = v;
      } else if (epi == EPI_GELU) {
        v = 0.5f * v * (1.0f + erff(v * 0.70710678118654752f));
        Out[(long)mm * ldo + cn] = v;
      } else if (epi == EPI_RES_ADD) {
        const long idx = (long)mm * ldo + cn;
        Out[idx] = Out[idx] + v;
      } else { // EPI_RES_WIN: window-token row -> [B,H,W,C] pixel, residual add
        const int N   = wh * ww;
        const int loc = mm % N;
        const int win = mm / N;
        const int ix = win % nWw; const int t2 = win / nWw;
        const int iy = t2 % nWh;  const int bb = t2 / nWh;
        const int rr = loc / ww,  cc = loc % ww;
        const int y = iy * wh + rr, x = ix * ww + cc;
        const long idx = (((long)bb * H_ + y) * W_ + x) * C_ + cn;
        Out[idx] = Out[idx] + v;
      }
    }
  }
}

// ---------------------------------------------------------------------------
// LayerNorm over C=256, one wave (32 lanes) per token. wave32 shfl reduction.
// ---------------------------------------------------------------------------
__global__ __launch_bounds__(256) void ln_kernel(
    const float* __restrict__ X, const float* __restrict__ g,
    const float* __restrict__ b, float* __restrict__ Out, int tokens)
{
  const int warp = threadIdx.x >> 5;
  const int lane = threadIdx.x & 31;
  const int t = blockIdx.x * 8 + warp;
  if (t >= tokens) return;
  const float* row = X + (long)t * C_;
  float s = 0.f, s2 = 0.f, vals[8];
  #pragma unroll
  for (int j = 0; j < 8; ++j) {
    float v = row[j * 32 + lane];
    vals[j] = v; s += v; s2 += v * v;
  }
  #pragma unroll
  for (int off = 16; off; off >>= 1) {
    s  += __shfl_xor(s,  off, 32);
    s2 += __shfl_xor(s2, off, 32);
  }
  const float mu  = s  * (1.0f / C_);
  const float var = s2 * (1.0f / C_) - mu * mu;
  const float rs  = rsqrtf(var + 1e-5f);
  float* orow = Out + (long)t * C_;
  #pragma unroll
  for (int j = 0; j < 8; ++j) {
    const int ch = j * 32 + lane;
    orow[ch] = (vals[j] - mu) * rs * g[ch] + b[ch];
  }
}

// ---------------------------------------------------------------------------
// Window partition of LN output + pos embed add. xw = xn(window order),
// qk = xw + pos.  pos is channel-first [B,C,H,W].
// ---------------------------------------------------------------------------
__global__ __launch_bounds__(256) void prep_qkv_input(
    const float* __restrict__ xn, const float* __restrict__ pos,
    float* __restrict__ xw, float* __restrict__ qk,
    int wh, int ww, int nWh, int nWw)
{
  const long i = (long)blockIdx.x * 256 + threadIdx.x;   // TOK_*64 float4 slots
  if (i >= (long)TOK_ * 64) return;
  const int t  = (int)(i >> 6);
  const int ch = ((int)i & 63) << 2;
  const int N = wh * ww;
  const int loc = t % N, win = t / N;
  const int ix = win % nWw; const int t2 = win / nWw;
  const int iy = t2 % nWh;  const int bb = t2 / nWh;
  const int rr = loc / ww,  cc = loc % ww;
  const int y = iy * wh + rr, x = ix * ww + cc;
  const long pix = ((long)bb * H_ + y) * W_ + x;
  float4 v = *(const float4*)(xn + pix * C_ + ch);
  const long HW = (long)H_ * W_;
  const long pbase = (((long)bb * C_ + ch) * H_ + y) * W_ + x;
  float4 p;
  p.x = pos[pbase]; p.y = pos[pbase + HW]; p.z = pos[pbase + 2 * HW]; p.w = pos[pbase + 3 * HW];
  *(float4*)(xw + (long)t * C_ + ch) = v;
  float4 s = make_float4(v.x + p.x, v.y + p.y, v.z + p.z, v.w + p.w);
  *(float4*)(qk + (long)t * C_ + ch) = s;
}

// ---------------------------------------------------------------------------
// Windowed multi-head attention, one workgroup per (window, head).
// K/V staged into LDS (f32) with GLOBAL_LOAD_ASYNC_TO_LDS_B128 (ASYNCcnt),
// per-wave 16-row query tiles; WMMA for QK^T and PV; full-row softmax in f32.
// Dynamic LDS = 516*N bytes (258 KB at N=512, within 320 KB/WGP).
// ---------------------------------------------------------------------------
__global__ __launch_bounds__(128) void attn_kernel(
    const float* __restrict__ qkv, const unsigned char* __restrict__ mask,
    float* __restrict__ aout, int N, int wh, int ww, int nWh, int nWw)
{
  extern __shared__ char smem[];
  float* Kl      = (float*)smem;                 // [N][32] f32
  float* Vl      = Kl + (long)N * HDIM_;         // [N][32] f32
  float* maskAdd = Vl + (long)N * HDIM_;         // [N]
  float* scores  = maskAdd + N;                  // [4 waves][16][N]

  const int lane = threadIdx.x & 31;
  const int warp = threadIdx.x >> 5;
  const int head = blockIdx.x % NHEAD_;
  const int win  = blockIdx.x / NHEAD_;
  const long base = (long)win * N;

#if USE_ASYNC_LDS
  // ---- K/V stage: per-lane async b128 copies, tracked by ASYNCcnt ----
  for (int i = threadIdx.x; i < N * 8; i += 128) {
    const int t = i >> 3, q = (i & 7) << 2;      // token, dword-quad within row
    const float* gk = qkv + (base + t) * (3 * C_) + C_ + head * HDIM_ + q;
    const float* gv = gk + C_;
    unsigned koff = (unsigned)(uintptr_t)(Kl + t * HDIM_ + q);
    unsigned voff = (unsigned)(uintptr_t)(Vl + t * HDIM_ + q);
    asm volatile("global_load_async_to_lds_b128 %0, %1, off"
                 :: "v"(koff), "v"((unsigned long long)(uintptr_t)gk) : "memory");
    asm volatile("global_load_async_to_lds_b128 %0, %1, off"
                 :: "v"(voff), "v"((unsigned long long)(uintptr_t)gv) : "memory");
  }
  for (int t = threadIdx.x; t < N; t += 128) {
    const int ix = win % nWw; const int t2 = win / nWw;
    const int iy = t2 % nWh;  const int bb = t2 / nWh;
    const int rr = t / ww,    cc = t % ww;
    const int y = iy * wh + rr, x = ix * ww + cc;
    maskAdd[t] = mask[((long)bb * H_ + y) * W_ + x] ? -1e9f : 0.0f;
  }
  asm volatile("s_wait_asynccnt 0" ::: "memory");
#else
  for (int t = threadIdx.x; t < N; t += 128) {
    const float* kp = qkv + (base + t) * (3 * C_) + C_ + head * HDIM_;
    const float* vp = kp + C_;
    #pragma unroll
    for (int d = 0; d < HDIM_; ++d) {
      Kl[t * HDIM_ + d] = kp[d];
      Vl[t * HDIM_ + d] = vp[d];
    }
    const int ix = win % nWw; const int t2 = win / nWw;
    const int iy = t2 % nWh;  const int bb = t2 / nWh;
    const int rr = t / ww,    cc = t % ww;
    const int y = iy * wh + rr, x = ix * ww + cc;
    maskAdd[t] = mask[((long)bb * H_ + y) * W_ + x] ? -1e9f : 0.0f;
  }
#endif
  __syncthreads();

  const int m  = lane & 15;
  const int hx = lane >> 4;
  float* myS = scores + (long)warp * 16 * N;
  const int qTiles = N >> 4;                       // 32 or 8, both divisible by 4
  const float qscale = 0.17677669529663687f;       // 1/sqrt(32)

  for (int mt = warp; mt < qTiles; mt += 4) {
    // Q tile as WMMA A operand (f16, scaled)
    const float* qrow = qkv + (base + mt * 16 + m) * (3 * C_) + head * HDIM_ + hx * 16;
    float4 q0 = *(const float4*)(qrow);
    float4 q1 = *(const float4*)(qrow + 4);
    float4 q2 = *(const float4*)(qrow + 8);
    float4 q3 = *(const float4*)(qrow + 12);
    v16h qa;
    qa[0]=(_Float16)(q0.x*qscale);  qa[1]=(_Float16)(q0.y*qscale);
    qa[2]=(_Float16)(q0.z*qscale);  qa[3]=(_Float16)(q0.w*qscale);
    qa[4]=(_Float16)(q1.x*qscale);  qa[5]=(_Float16)(q1.y*qscale);
    qa[6]=(_Float16)(q1.z*qscale);  qa[7]=(_Float16)(q1.w*qscale);
    qa[8]=(_Float16)(q2.x*qscale);  qa[9]=(_Float16)(q2.y*qscale);
    qa[10]=(_Float16)(q2.z*qscale); qa[11]=(_Float16)(q2.w*qscale);
    qa[12]=(_Float16)(q3.x*qscale); qa[13]=(_Float16)(q3.y*qscale);
    qa[14]=(_Float16)(q3.z*qscale); qa[15]=(_Float16)(q3.w*qscale);

    // S[16,N] = Q * K^T, one WMMA per 16-key tile (K-dim = head dim = 32)
    for (int jt = 0; jt < qTiles; ++jt) {
      const float* kr = Kl + (jt * 16 + m) * HDIM_ + hx * 16;   // B: col n=m
      float4 k0 = *(const float4*)(kr);
      float4 k1 = *(const float4*)(kr + 4);
      float4 k2 = *(const float4*)(kr + 8);
      float4 k3 = *(const float4*)(kr + 12);
      v16h kb = cvt16(k0, k1, k2, k3);
      v8f c = {};
      c = wmma_f16(qa, kb, c);
      const float madd = maskAdd[jt * 16 + m];
      #pragma unroll
      for (int r = 0; r < 8; ++r)
        myS[(r + 8 * hx) * N + jt * 16 + m] = c[r] + madd;
    }
    __syncthreads();

    // row softmax (lanes 0..15, one row each)
    if (lane < 16) {
      float* srow = myS + lane * N;
      float mx = -3.0e38f;
      for (int j = 0; j < N; ++j) mx = fmaxf(mx, srow[j]);
      float sum = 0.f;
      for (int j = 0; j < N; ++j) { float e = __expf(srow[j] - mx); srow[j] = e; sum += e; }
      const float inv = 1.0f / sum;
      for (int j = 0; j < N; ++j) srow[j] *= inv;
    }
    __syncthreads();

    // O[16,32] = P * V  (two 16-col output tiles over head dim)
    #pragma unroll
    for (int ot = 0; ot < 2; ++ot) {
      v8f acc = {};
      for (int kc = 0; kc < (N >> 5); ++kc) {
        v16h pa;
        const float* prow = myS + m * N + kc * 32 + hx * 16;
        #pragma unroll
        for (int s = 0; s < 8; ++s)  pa[s] = (_Float16)prow[s];
        #pragma unroll
        for (int s = 8; s < 16; ++s) pa[s] = (_Float16)prow[s];
        v16h vb;
        const float* vcol = Vl + (kc * 32 + hx * 16) * HDIM_ + ot * 16 + m;
        #pragma unroll
        for (int s = 0; s < 16; ++s) vb[s] = (_Float16)vcol[s * HDIM_];
        acc = wmma_f16(pa, vb, acc);
      }
      #pragma unroll
      for (int r = 0; r < 8; ++r)
        aout[(base + mt * 16 + r + 8 * hx) * C_ + head * HDIM_ + ot * 16 + m] = acc[r];
    }
  }
}

// ---------------------------------------------------------------------------
// Layout transposes
// ---------------------------------------------------------------------------
__global__ __launch_bounds__(256) void chw2hwc(const float* __restrict__ src,
                                               float* __restrict__ dst)
{
  const long i = (long)blockIdx.x * 256 + threadIdx.x;     // indexes [B,H,W,C]
  if (i >= (long)TOK_ * C_) return;
  const int ch = (int)(i % C_);
  const long pix = i / C_;
  const int x = (int)(pix % W_);
  const long t = pix / W_;
  const int y = (int)(t % H_);
  const int bb = (int)(t / H_);
  dst[i] = src[(((long)bb * C_ + ch) * H_ + y) * W_ + x];
}

__global__ __launch_bounds__(256) void hwc2chw(const float* __restrict__ src,
                                               float* __restrict__ dst)
{
  const long i = (long)blockIdx.x * 256 + threadIdx.x;     // indexes [B,C,H,W]
  if (i >= (long)TOK_ * C_) return;
  const int x = (int)(i % W_);
  const long t1 = i / W_;
  const int y = (int)(t1 % H_);
  const long t2 = t1 / H_;
  const int ch = (int)(t2 % C_);
  const int bb = (int)(t2 / C_);
  dst[i] = src[(((long)bb * H_ + y) * W_ + x) * C_ + ch];
}

// ---------------------------------------------------------------------------
extern "C" void kernel_launch(void* const* d_in, const int* in_sizes, int n_in,
                              void* d_out, int out_size, void* d_ws, size_t ws_size,
                              hipStream_t stream)
{
  (void)in_sizes; (void)n_in; (void)out_size; (void)ws_size;

  const float* src   = (const float*)d_in[0];
  const float* pos4  = (const float*)d_in[1];
  const float* pos8  = (const float*)d_in[2];
  const unsigned char* mask4 = (const unsigned char*)d_in[3];
  const unsigned char* mask8 = (const unsigned char*)d_in[4];
  const float* ln1_g = (const float*)d_in[5];
  const float* ln1_b = (const float*)d_in[6];
  const float* qkv_w = (const float*)d_in[7];
  const float* qkv_b = (const float*)d_in[8];
  const float* out_w = (const float*)d_in[9];
  const float* out_b = (const float*)d_in[10];
  const float* ln2_g = (const float*)d_in[11];
  const float* ln2_b = (const float*)d_in[12];
  const float* fc1_w = (const float*)d_in[13];
  const float* fc1_b = (const float*)d_in[14];
  const float* fc2_w = (const float*)d_in[15];
  const float* fc2_b = (const float*)d_in[16];

  float* ws   = (float*)d_ws;
  float* xh   = ws;                          // [TOK,C] channel-last activation
  float* xn   = xh   + (long)TOK_ * C_;      // LN scratch
  float* xw   = xn   + (long)TOK_ * C_;      // windowed x
  float* qk   = xw   + (long)TOK_ * C_;      // windowed x+pos
  float* aout = qk   + (long)TOK_ * C_;      // attention output (windowed)
  float* qkv  = aout + (long)TOK_ * C_;      // [TOK,768]; reused as h1 [TOK,512]
  float* h1   = qkv;

  const int ewin[L_][2] = {{32,16},{16,8},{32,16},{16,8}};

  // allow large dynamic LDS for the attention kernel (N=512 -> 258 KB)
  (void)hipFuncSetAttribute(reinterpret_cast<const void*>(attn_kernel),
                            hipFuncAttributeMaxDynamicSharedMemorySize, 516 * 512);

  const int ELEMS = TOK_ * C_;
  chw2hwc<<<(ELEMS + 255) / 256, 256, 0, stream>>>(src, xh);

  for (int i = 0; i < L_; ++i) {
    const int wh = ewin[i][0], ww = ewin[i][1];
    const int nWh = H_ / wh, nWw = W_ / ww;
    const int N = wh * ww;
    const float* pos = (i % 2 == 0) ? pos4 : pos8;
    const unsigned char* mask = (i % 2 == 0) ? mask4 : mask8;

    // LN1 + window partition (+pos)
    ln_kernel<<<TOK_ / 8, 256, 0, stream>>>(xh, ln1_g + i * C_, ln1_b + i * C_, xn, TOK_);
    prep_qkv_input<<<(TOK_ * 64 + 255) / 256, 256, 0, stream>>>(xn, pos, xw, qk,
                                                                wh, ww, nWh, nWw);

    // Q,K projection from (x+pos); V projection from x — into [TOK,768]
    gemm64_kernel<<<(TOK_ / 64) * (2 * C_ / 64), 128, 0, stream>>>(
        qk, C_, qkv_w + (long)i * 3 * C_ * C_, C_,
        qkv_b + (long)i * 3 * C_, qkv, 3 * C_,
        TOK_, 2 * C_, C_, EPI_BIAS, 0, 0, 0, 0);
    gemm64_kernel<<<(TOK_ / 64) * (C_ / 64), 128, 0, stream>>>(
        xw, C_, qkv_w + (long)i * 3 * C_ * C_ + (long)2 * C_ * C_, C_,
        qkv_b + (long)i * 3 * C_ + 2 * C_, qkv + 2 * C_, 3 * C_,
        TOK_, C_, C_, EPI_BIAS, 0, 0, 0, 0);

    // attention per (window, head)
    {
      const int Bn = B_ * nWh * nWw;
      const size_t shmem = (size_t)516 * N;
      attn_kernel<<<Bn * NHEAD_, 128, shmem, stream>>>(qkv, mask, aout,
                                                       N, wh, ww, nWh, nWw);
    }

    // out-proj + window_reverse + residual into xh
    gemm64_kernel<<<(TOK_ / 64) * (C_ / 64), 128, 0, stream>>>(
        aout, C_, out_w + (long)i * C_ * C_, C_,
        out_b + (long)i * C_, xh, C_,
        TOK_, C_, C_, EPI_RES_WIN, wh, ww, nWh, nWw);

    // LN2 + MLP (fc1+GELU, fc2+residual)
    ln_kernel<<<TOK_ / 8, 256, 0, stream>>>(xh, ln2_g + i * C_, ln2_b + i * C_, xn, TOK_);
    gemm64_kernel<<<(TOK_ / 64) * (DFF_ / 64), 128, 0, stream>>>(
        xn, C_, fc1_w + (long)i * DFF_ * C_, C_,
        fc1_b + (long)i * DFF_, h1, DFF_,
        TOK_, DFF_, C_, EPI_GELU, 0, 0, 0, 0);
    gemm64_kernel<<<(TOK_ / 64) * (C_ / 64), 128, 0, stream>>>(
        h1, DFF_, fc2_w + (long)i * C_ * DFF_, DFF_,
        fc2_b + (long)i * C_, xh, C_,
        TOK_, C_, DFF_, EPI_RES_ADD, 0, 0, 0, 0);
  }

  hwc2chw<<<(ELEMS + 255) / 256, 256, 0, stream>>>(xh, (float*)d_out);
}
